// SA_MultiHeadAttention_46866683134021
// MI455X (gfx1250) — matrix operand order repaired and checked by reading, not compiled, not run on hardware
//
#include <hip/hip_runtime.h>
#include <stdint.h>

// Problem constants (from the reference): B=8, T=2048, C=1024, H=16, D=64
#define BB   8
#define TT   2048
#define CC   1024
#define HH   16
#define DDIM 64
#define MTOT (BB * TT)  // 16384 rows

typedef __attribute__((ext_vector_type(16))) __bf16         v16bf;
typedef __attribute__((ext_vector_type(16))) unsigned short ushort16;
typedef __attribute__((ext_vector_type(8)))  unsigned short ushort8;
typedef __attribute__((ext_vector_type(8)))  float          v8f;
typedef __attribute__((ext_vector_type(4)))  float          v4f;
typedef __attribute__((ext_vector_type(4)))  unsigned short u16x4;

static __device__ __forceinline__ __bf16 f2bf(float f) {
  // round-to-nearest-even f32 -> bf16
  uint32_t u = __builtin_bit_cast(uint32_t, f);
  uint32_t r = (u + 0x7FFFu + ((u >> 16) & 1u)) >> 16;
  unsigned short s = (unsigned short)r;
  return __builtin_bit_cast(__bf16, s);
}

static __device__ __forceinline__ v16bf mk_frag(ushort8 lo, ushort8 hi) {
  ushort16 w;
#pragma unroll
  for (int i = 0; i < 8; ++i) { w[i] = lo[i]; w[i + 8] = hi[i]; }
  return __builtin_bit_cast(v16bf, w);
}

// A-fragment (16x32 bf16) from row-major [M][ld] memory.
// lane m = lane&15, half = lane>>4 holds K = {8h..8h+7} U {16+8h..16+8h+7}.
static __device__ __forceinline__ v16bf load_a(const __bf16* base, int m0, int k0,
                                               int ld, int lane) {
  int m = lane & 15, h = lane >> 4;
  const __bf16* p = base + (size_t)(m0 + m) * ld + k0 + 8 * h;
  ushort8 lo = *(const ushort8*)(p);
  ushort8 hi = *(const ushort8*)(p + 16);
  return mk_frag(lo, hi);
}

// B-fragment (32x16 bf16) from n-major [N][ld] memory (K contiguous per lane).
// lane n = lane&15, half = lane>>4 holds K = {8h..8h+7} U {16+8h..16+8h+7}.
static __device__ __forceinline__ v16bf load_b(const __bf16* base, int n0, int k0,
                                               int ld, int lane) {
  int n = lane & 15, h = lane >> 4;
  const __bf16* p = base + (size_t)(n0 + n) * ld + k0 + 8 * h;
  ushort8 lo = *(const ushort8*)(p);
  ushort8 hi = *(const ushort8*)(p + 16);
  return mk_frag(lo, hi);
}

static __device__ __forceinline__ v8f wmma_bf16(v16bf a, v16bf b, v8f c) {
  return __builtin_amdgcn_wmma_f32_16x16x32_bf16(false, a, false, b, (short)0, c,
                                                 false, false);
}

// ---------------------------------------------------------------- converts
__global__ void k_f32_to_bf16(const float* __restrict__ in, __bf16* __restrict__ out,
                              int n4) {
  int i = blockIdx.x * blockDim.x + threadIdx.x;
  if (i < n4) {
    v4f f = *(const v4f*)(in + (size_t)i * 4);
    u16x4 o;
#pragma unroll
    for (int j = 0; j < 4; ++j) o[j] = __builtin_bit_cast(unsigned short, f2bf(f[j]));
    *(u16x4*)((unsigned short*)out + (size_t)i * 4) = o;
  }
}

// src [batch][rows][cols] f32  ->  dst [batch][cols][rows] bf16
__global__ void k_transpose_bf16(const float* __restrict__ src,
                                 __bf16* __restrict__ dst, int rows, int cols) {
  int i = blockIdx.x * blockDim.x + threadIdx.x;  // linear index into dst
  int per = rows * cols;
  int b = i / per, r2 = i % per;
  int c = r2 / rows, r = r2 % rows;
  dst[i] = f2bf(src[(size_t)b * per + (size_t)r * cols + c]);
}

// ------------------------------------------------------------- QKV GEMM
// out[b,h,t,d] (or transposed [b,h,d,t] for V) = sum_c x[b,t,c] * W[h,c,d]
// xb: [MTOT][CC] bf16; wt: [HH][DDIM][CC] bf16 (n-major per head)
__global__ void __launch_bounds__(256)
k_qkv_gemm(const __bf16* __restrict__ xb, const __bf16* __restrict__ wt,
           __bf16* __restrict__ out, float scale, int vtranspose) {
  __shared__ __align__(16) __bf16 Vtile[8][16][72];  // per-wave 16x64 tile (+pad)
  int lane = threadIdx.x & 31;
  int wave = threadIdx.x >> 5;
  int h    = blockIdx.y;
  int m0   = blockIdx.x * 128 + wave * 16;
  const __bf16* w_h = wt + (size_t)h * DDIM * CC;

  v8f acc[4] = {};
  // Ping-pong double buffering: two fragment sets alternate roles; each set is
  // refilled directly by loads (no register copies). Final iteration over-reads
  // one k-tile; the workspace layout makes that safe. unroll(1) pins the
  // pipeline depth so the compiler doesn't re-unroll and spill-rotate regs.
  v16bf a0 = load_a(xb, m0, 0, CC, lane);
  v16bf b0[4];
#pragma unroll
  for (int dt = 0; dt < 4; ++dt) b0[dt] = load_b(w_h, dt * 16, 0, CC, lane);

#pragma unroll 1
  for (int k0 = 0; k0 < CC; k0 += 64) {
    v16bf a1 = load_a(xb, m0, k0 + 32, CC, lane);
    v16bf b1[4];
#pragma unroll
    for (int dt = 0; dt < 4; ++dt) b1[dt] = load_b(w_h, dt * 16, k0 + 32, CC, lane);
#pragma unroll
    for (int dt = 0; dt < 4; ++dt) acc[dt] = wmma_bf16(a0, b0[dt], acc[dt]);

    a0 = load_a(xb, m0, k0 + 64, CC, lane);
#pragma unroll
    for (int dt = 0; dt < 4; ++dt) b0[dt] = load_b(w_h, dt * 16, k0 + 64, CC, lane);
#pragma unroll
    for (int dt = 0; dt < 4; ++dt) acc[dt] = wmma_bf16(a1, b1[dt], acc[dt]);
  }

  int n = lane & 15, half = lane >> 4;
  int bidx = m0 >> 11;          // batch (16-row tile never straddles a batch)
  int t0   = m0 & (TT - 1);

  if (vtranspose) {
    // stage C tile in LDS, then write Vt[d][t] as 32B contiguous runs per lane
#pragma unroll
    for (int dt = 0; dt < 4; ++dt)
#pragma unroll
      for (int i = 0; i < 8; ++i)
        Vtile[wave][i + 8 * half][dt * 16 + n] = f2bf(acc[dt][i] * scale);
    size_t base = (size_t)(bidx * HH + h) * DDIM * TT;
#pragma unroll
    for (int r = 0; r < 2; ++r) {
      int d = lane * 2 + r;
      ushort16 w;
#pragma unroll
      for (int t = 0; t < 16; ++t)
        w[t] = __builtin_bit_cast(unsigned short, Vtile[wave][t][d]);
      *(ushort16*)((unsigned short*)out + base + (size_t)d * TT + t0) = w;
    }
  } else {
#pragma unroll
    for (int dt = 0; dt < 4; ++dt) {
#pragma unroll
      for (int i = 0; i < 8; ++i) {
        int t = t0 + i + 8 * half;
        int d = dt * 16 + n;
        out[((size_t)(bidx * HH + h) * TT + t) * DDIM + d] = f2bf(acc[dt][i] * scale);
      }
    }
  }
}

// ---------------------------------------------------------- flash attention
// Q,K: [B*H][T][D] bf16 (Q pre-scaled by 1/sqrt(D)); Vt: [B*H][D][T] bf16
// att: [B*T][H*D] bf16
__global__ void __launch_bounds__(256)
k_attn(const __bf16* __restrict__ Q, const __bf16* __restrict__ K,
       const __bf16* __restrict__ Vt, __bf16* __restrict__ att) {
  __shared__ __align__(16) __bf16 Plds[8][16][40];  // per-wave 16x32 P tile (+pad)
  int lane = threadIdx.x & 31;
  int wave = threadIdx.x >> 5;
  int task = blockIdx.x * 8 + wave;     // one wave = one (bh, 16-row q block)
  const int QB = TT / 16;               // 128 query blocks
  int bh = task / QB;
  int q0 = (task % QB) * 16;
  int b  = bh / HH, h = bh % HH;

  const __bf16* q_p = Q  + (size_t)bh * TT * DDIM;
  const __bf16* k_p = K  + (size_t)bh * TT * DDIM;
  const __bf16* v_p = Vt + (size_t)bh * DDIM * TT;

  v16bf qf0 = load_a(q_p, q0, 0, DDIM, lane);
  v16bf qf1 = load_a(q_p, q0, 32, DDIM, lane);

  v8f o[4] = {};
  float rmax[8], rsum[8];
#pragma unroll
  for (int i = 0; i < 8; ++i) { rmax[i] = -__builtin_inff(); rsum[i] = 0.f; }

  int n = lane & 15, half = lane >> 4;

#pragma unroll 1
  for (int s0 = 0; s0 < q0 + 16; s0 += 32) {  // causal: only key blocks <= q rows
    // hoist ALL loads: K frags for the score WMMAs, V frags for the PV WMMAs.
    // V loads are independent of the softmax math and overlap it.
    v16bf kf00 = load_b(k_p, s0,      0,  DDIM, lane);
    v16bf kf01 = load_b(k_p, s0,      32, DDIM, lane);
    v16bf kf10 = load_b(k_p, s0 + 16, 0,  DDIM, lane);
    v16bf kf11 = load_b(k_p, s0 + 16, 32, DDIM, lane);
    v16bf vf[4];
#pragma unroll
    for (int dt = 0; dt < 4; ++dt) vf[dt] = load_b(v_p, dt * 16, s0, TT, lane);

    v8f sacc[2] = {};
    sacc[0] = wmma_bf16(qf0, kf00, sacc[0]);
    sacc[0] = wmma_bf16(qf1, kf01, sacc[0]);
    sacc[1] = wmma_bf16(qf0, kf10, sacc[1]);
    sacc[1] = wmma_bf16(qf1, kf11, sacc[1]);

    // causal mask + running row max (rows live across 16 lanes of a half-wave)
    float nmax[8];
#pragma unroll
    for (int i = 0; i < 8; ++i) {
      int row = q0 + i + 8 * half;
      float s0v = ((s0 + n) <= row) ? sacc[0][i] : -__builtin_inff();
      float s1v = ((s0 + 16 + n) <= row) ? sacc[1][i] : -__builtin_inff();
      sacc[0][i] = s0v;
      sacc[1][i] = s1v;
      float m = fmaxf(s0v, s1v);
#pragma unroll
      for (int x = 1; x < 16; x <<= 1) m = fmaxf(m, __shfl_xor(m, x, 32));
      nmax[i] = fmaxf(rmax[i], m);
    }
#pragma unroll
    for (int i = 0; i < 8; ++i) {
      float p0 = __expf(sacc[0][i] - nmax[i]);
      float p1 = __expf(sacc[1][i] - nmax[i]);
      float ls = p0 + p1;
#pragma unroll
      for (int x = 1; x < 16; x <<= 1) ls += __shfl_xor(ls, x, 32);
      float corr = __expf(rmax[i] - nmax[i]);
      rsum[i] = rsum[i] * corr + ls;
      rmax[i] = nmax[i];
#pragma unroll
      for (int dt = 0; dt < 4; ++dt) o[dt][i] *= corr;
      // stage P (C layout -> LDS row-major) for the A-fragment reload
      Plds[wave][i + 8 * half][n]      = f2bf(p0);
      Plds[wave][i + 8 * half][16 + n] = f2bf(p1);
    }
    {
      const __bf16* pr = &Plds[wave][lane & 15][0];
      ushort8 lo = *(const ushort8*)(pr + 8 * half);
      ushort8 hi = *(const ushort8*)(pr + 16 + 8 * half);
      v16bf pf = mk_frag(lo, hi);
#pragma unroll
      for (int dt = 0; dt < 4; ++dt) o[dt] = wmma_bf16(pf, vf[dt], o[dt]);
    }
  }
  // normalize and store att[b*T + t][h*64 + d]
#pragma unroll
  for (int i = 0; i < 8; ++i) {
    float inv = 1.0f / rsum[i];
    int t = q0 + i + 8 * half;
    size_t rowoff = ((size_t)b * TT + t) * (HH * DDIM) + h * DDIM;
#pragma unroll
    for (int dt = 0; dt < 4; ++dt)
      att[rowoff + dt * 16 + n] = f2bf(o[dt][i] * inv);
  }
}

// ------------------------------------------------------------ output GEMM
// ab: [MTOT][1024] bf16; wot: [CC][1024] bf16 (n-major); out f32 + bias
__global__ void __launch_bounds__(256)
k_out_gemm(const __bf16* __restrict__ ab, const __bf16* __restrict__ wot,
           const float* __restrict__ bo, float* __restrict__ out) {
  int lane = threadIdx.x & 31;
  int wave = threadIdx.x >> 5;
  int m0   = blockIdx.x * 128 + wave * 16;
  int n0b  = blockIdx.y * 64;
  const int KK = HH * DDIM;  // 1024
  v8f acc[4] = {};

  v16bf a0 = load_a(ab, m0, 0, KK, lane);
  v16bf b0[4];
#pragma unroll
  for (int dt = 0; dt < 4; ++dt) b0[dt] = load_b(wot, n0b + dt * 16, 0, KK, lane);

#pragma unroll 1
  for (int k0 = 0; k0 < KK; k0 += 64) {
    v16bf a1 = load_a(ab, m0, k0 + 32, KK, lane);
    v16bf b1[4];
#pragma unroll
    for (int dt = 0; dt < 4; ++dt)
      b1[dt] = load_b(wot, n0b + dt * 16, k0 + 32, KK, lane);
#pragma unroll
    for (int dt = 0; dt < 4; ++dt) acc[dt] = wmma_bf16(a0, b0[dt], acc[dt]);

    a0 = load_a(ab, m0, k0 + 64, KK, lane);
#pragma unroll
    for (int dt = 0; dt < 4; ++dt)
      b0[dt] = load_b(wot, n0b + dt * 16, k0 + 64, KK, lane);
#pragma unroll
    for (int dt = 0; dt < 4; ++dt) acc[dt] = wmma_bf16(a1, b1[dt], acc[dt]);
  }

  int n = lane & 15, half = lane >> 4;
#pragma unroll
  for (int dt = 0; dt < 4; ++dt) {
    int col = n0b + dt * 16 + n;
    float bias = bo[col];
#pragma unroll
    for (int i = 0; i < 8; ++i) {
      int m = m0 + i + 8 * half;
      out[(size_t)m * CC + col] = acc[dt][i] + bias;
    }
  }
}

// ---------------------------------------------------------------- launcher
extern "C" void kernel_launch(void* const* d_in, const int* in_sizes, int n_in,
                              void* d_out, int out_size, void* d_ws, size_t ws_size,
                              hipStream_t stream) {
  const float* x  = (const float*)d_in[0];
  const float* Wq = (const float*)d_in[1];
  const float* Wk = (const float*)d_in[2];
  const float* Wv = (const float*)d_in[3];
  const float* Wo = (const float*)d_in[4];
  const float* bo = (const float*)d_in[5];
  float* out = (float*)d_out;

  // Workspace layout (bf16 elements), ~168 MB total.
  // Order matters: buffers that GEMMs may over-read by one k-tile (xb, wtq/k/v,
  // wot, ab) are each followed by another workspace buffer; vt (never over-read)
  // is last.
  __bf16* p = (__bf16*)d_ws;
  __bf16* xb  = p; p += (size_t)MTOT * CC;            // x in bf16
  __bf16* wtq = p; p += (size_t)HH * DDIM * CC;       // Wq^T per head [H][D][C]
  __bf16* wtk = p; p += (size_t)HH * DDIM * CC;
  __bf16* wtv = p; p += (size_t)HH * DDIM * CC;
  __bf16* wot = p; p += (size_t)CC * HH * DDIM;       // Wo^T [C][HD]
  __bf16* ab  = p; p += (size_t)MTOT * HH * DDIM;     // concat heads [BT][HD]
  __bf16* qb  = p; p += (size_t)BB * HH * TT * DDIM;  // Q [BH][T][D], pre-scaled
  __bf16* kb  = p; p += (size_t)BB * HH * TT * DDIM;  // K [BH][T][D]
  __bf16* vt  = p; p += (size_t)BB * HH * TT * DDIM;  // V^T [BH][D][T]

  const int nx = MTOT * CC;          // 16,777,216
  const int nw = HH * DDIM * CC;     // 1,048,576

  k_f32_to_bf16<<<dim3(nx / 4 / 256), dim3(256), 0, stream>>>(x, xb, nx / 4);
  k_transpose_bf16<<<dim3(nw / 256), dim3(256), 0, stream>>>(Wq, wtq, CC, DDIM);
  k_transpose_bf16<<<dim3(nw / 256), dim3(256), 0, stream>>>(Wk, wtk, CC, DDIM);
  k_transpose_bf16<<<dim3(nw / 256), dim3(256), 0, stream>>>(Wv, wtv, CC, DDIM);
  k_transpose_bf16<<<dim3(nw / 256), dim3(256), 0, stream>>>(Wo, wot, HH * DDIM, CC);

  k_qkv_gemm<<<dim3(MTOT / 128, HH), dim3(256), 0, stream>>>(xb, wtq, qb, 0.125f, 0);
  k_qkv_gemm<<<dim3(MTOT / 128, HH), dim3(256), 0, stream>>>(xb, wtk, kb, 1.0f, 0);
  k_qkv_gemm<<<dim3(MTOT / 128, HH), dim3(256), 0, stream>>>(xb, wtv, vt, 1.0f, 1);

  k_attn<<<dim3(BB * HH * (TT / 16) / 8), dim3(256), 0, stream>>>(qb, kb, vt, ab);

  k_out_gemm<<<dim3(MTOT / 128, CC / 64), dim3(256), 0, stream>>>(ab, wot, bo, out);
}
